// VGG11_57397942944391
// MI455X (gfx1250) — 16-bit and 8-bit matrix operand order repaired and checked by reading, compiled, not run on hardware
//
#include <hip/hip_runtime.h>
#include <cstddef>

typedef __attribute__((ext_vector_type(8)))  int      v8i;
typedef __attribute__((ext_vector_type(8)))  float    v8f;
typedef __attribute__((ext_vector_type(16))) _Float16 v16h;
typedef __attribute__((ext_vector_type(4)))  unsigned uv4;
typedef __attribute__((ext_vector_type(4)))  int      iv4;
typedef __attribute__((ext_vector_type(8)))  int      iv8;

union V8IB  { v8i  v; int i[8]; };
union V16HB { v16h v; unsigned u[8]; };

#if defined(__HIP_DEVICE_COMPILE__) && defined(__has_builtin)
#if __has_builtin(__builtin_amdgcn_tensor_load_to_lds) && \
    __has_builtin(__builtin_amdgcn_s_wait_tensorcnt)
#define USE_TDM 1
#endif
#endif

#ifdef USE_TDM
typedef __attribute__((address_space(3))) signed char lds_char;
__device__ inline void tdm_copy_to_lds(const signed char* gsrc, signed char* ldst,
                                       unsigned bytes) {
  const unsigned long long gaddr = (unsigned long long)gsrc;
  const unsigned lds_off = (unsigned)(unsigned long long)(lds_char*)ldst;
  const unsigned td0 = bytes >> 3;
  uv4 g0;
  g0[0] = 1u;
  g0[1] = lds_off;
  g0[2] = (unsigned)(gaddr & 0xFFFFFFFFu);
  g0[3] = (unsigned)((gaddr >> 32) & 0x1FFFFFFu) | (2u << 30);
  iv8 g1;
  g1[0] = (int)(3u << 16);
  g1[1] = (int)((td0 & 0xFFFFu) << 16);
  g1[2] = (int)((td0 >> 16) | (1u << 16));
  g1[3] = (int)(td0 << 16);
  g1[4] = 1;
  g1[5] = (int)td0;
  g1[6] = 0;
  g1[7] = 0;
  iv4 g2 = {0, 0, 0, 0};
  iv4 g3 = {0, 0, 0, 0};
#if __clang_major__ >= 23
  iv8 g4 = {0, 0, 0, 0, 0, 0, 0, 0};
  __builtin_amdgcn_tensor_load_to_lds(g0, g1, g2, g3, g4, 0);
#else
  __builtin_amdgcn_tensor_load_to_lds(g0, g1, g2, g3, 0);
#endif
  __builtin_amdgcn_s_wait_tensorcnt(0);
}
#endif

__global__ void binarize_x_nhwc(const float* __restrict__ in,
                                signed char* __restrict__ out,
                                int N, int C, int H, int W) {
  const int total = N * C * H * W;
  for (int i = blockIdx.x * blockDim.x + threadIdx.x; i < total;
       i += gridDim.x * blockDim.x) {
    int c = i % C; int t = i / C;
    int x = t % W; t /= W;
    int y = t % H; int n = t / H;
    out[i] = (in[((size_t)(n * C + c) * H + y) * W + x] >= 0.0f)
                 ? (signed char)1 : (signed char)(-1);
  }
}

__global__ void binarize_w_reorder(const float* __restrict__ in,
                                   signed char* __restrict__ out,
                                   int Cout, int Cin) {
  const int total = Cout * Cin * 9;
  for (int i = blockIdx.x * blockDim.x + threadIdx.x; i < total;
       i += gridDim.x * blockDim.x) {
    int c = i % Cin; int t = i / Cin;
    int s = t % 9;   int o = t / 9;
    int ky = s / 3, kx = s - ky * 3;
    float v = in[((size_t)(o * Cin + c) * 3 + ky) * 3 + kx];
    out[i] = (v >= 0.0f) ? (signed char)1 : (signed char)(-1);
  }
}

__global__ void binarize_f32_i8(const float* __restrict__ in,
                                signed char* __restrict__ out, int n) {
  for (int i = blockIdx.x * blockDim.x + threadIdx.x; i < n;
       i += gridDim.x * blockDim.x)
    out[i] = (in[i] >= 0.0f) ? (signed char)1 : (signed char)(-1);
}

__global__ void pool2_i8_nhwc(const signed char* __restrict__ in,
                              signed char* __restrict__ out,
                              int N, int C, int H, int W) {
  const int Ho = H >> 1, Wo = W >> 1;
  const int total = N * Ho * Wo * C;
  const size_t rowC = (size_t)W * C;
  for (int i = blockIdx.x * blockDim.x + threadIdx.x; i < total;
       i += gridDim.x * blockDim.x) {
    int c  = i % C;  int t = i / C;
    int wo = t % Wo; t /= Wo;
    int ho = t % Ho; int n = t / Ho;
    const signed char* p =
        in + ((size_t)(n * H + 2 * ho) * W + 2 * wo) * C + c;
    int a0 = p[0], a1 = p[C], a2 = p[rowC], a3 = p[rowC + C];
    int m0 = a0 > a1 ? a0 : a1;
    int m1 = a2 > a3 ? a2 : a3;
    out[i] = (signed char)(m0 > m1 ? m0 : m1);
  }
}

template <int CIN>
__global__ void __launch_bounds__(128) conv_bin_wmma(
    const signed char* __restrict__ act, const signed char* __restrict__ wsg,
    signed char* __restrict__ out, int N, int H, int W, int Cout) {
  constexpr int Ktot = CIN * 9;
  extern __shared__ signed char Bsh[];
  const int lane = threadIdx.x & 31;
  const int wave = threadIdx.x >> 5;
  const int n0 = blockIdx.y << 4;

  const signed char* wpanel = wsg + (size_t)n0 * Ktot;
#ifdef USE_TDM
  if (wave == 0) tdm_copy_to_lds(wpanel, Bsh, 16u * Ktot);
#else
  {
    const int4* src = (const int4*)wpanel;
    int4* dst = (int4*)Bsh;
    for (int i = threadIdx.x; i < Ktot; i += 128) dst[i] = src[i];
  }
#endif
  __syncthreads();

  const int HW = H * W;
  const int mt = blockIdx.x * 4 + wave;
  const int rl = lane & 15;
  const int sel = lane >> 4;

  const int m = (mt << 4) + rl;
  const int nimg = m / HW;
  const int rem = m - nimg * HW;
  const int oy = rem / W;
  const int ox = rem - oy * W;
  const signed char* img = act + (size_t)nimg * HW * CIN;
  const signed char* bcol = Bsh + rl * Ktot;

  v8i acc = {};
  for (int k0 = 0; k0 < Ktot; k0 += 64) {
    __builtin_prefetch(img + ((size_t)(oy * W + ox)) * CIN + k0, 0, 0);
    V8IB A, B;
#pragma unroll
    for (int v = 0; v < 8; ++v) {
      const int ab = ((v >> 2) << 5) + (((v >> 1) & 1) << 4) + ((v & 1) << 2) + (sel << 3);
      const int k = k0 + ab;
      int aval = 0;
      if ((Ktot % 64 == 0) || (k < Ktot)) {
        const int seg = k / CIN;
        const int c0  = k - seg * CIN;
        const int ky  = seg / 3;
        const int kx  = seg - ky * 3;
        const int iy  = oy + ky - 1;
        const int ix  = ox + kx - 1;
        if ((unsigned)iy < (unsigned)H && (unsigned)ix < (unsigned)W)
          aval = *(const int*)(img + ((size_t)(iy * W + ix)) * CIN + c0);
      }
      A.i[v] = aval;
      const int bb = ((v >> 2) << 5) + (((v >> 1) & 1) << 4) + ((v & 1) << 2) + (sel << 3);
      const int kb = k0 + bb;
      B.i[v] = ((Ktot % 64 == 0) || (kb < Ktot)) ? *(const int*)(bcol + kb) : 0;
    }
    acc = __builtin_amdgcn_wmma_i32_16x16x64_iu8(true, A.v, true, B.v, acc,
                                                 false, false);
  }

  const int col = n0 + rl;
  const int mbase = (mt << 4) + (sel << 3);
#pragma unroll
  for (int r = 0; r < 8; ++r)
    out[(size_t)(mbase + r) * Cout + col] =
        (acc[r] >= 0) ? (signed char)1 : (signed char)(-1);
}

__global__ void __launch_bounds__(128) gemm_bin_wmma(
    const signed char* __restrict__ A8, const signed char* __restrict__ W8,
    void* __restrict__ outp, int M, int N, int K, int mode) {
  __shared__ signed char Wsh[16 * 512];
  const int lane = threadIdx.x & 31;
  const int wave = threadIdx.x >> 5;
  const int mt = blockIdx.x * 4 + wave;
  const int n0 = blockIdx.y << 4;
  {
    const int4* src = (const int4*)(W8 + (size_t)n0 * K);
    int4* dst = (int4*)Wsh;
    for (int i = threadIdx.x; i < (16 * K) / 16; i += 128) dst[i] = src[i];
  }
  __syncthreads();

  const int rl = lane & 15;
  const int sel = lane >> 4;
  const int row = (mt << 4) + rl;
  const signed char* bcol = Wsh + rl * K;

  v8i acc = {};
  for (int k0 = 0; k0 < K; k0 += 64) {
    V8IB A, B;
#pragma unroll
    for (int v = 0; v < 8; ++v) {
      const int ab = ((v >> 2) << 5) + (((v >> 1) & 1) << 4) + ((v & 1) << 2) + (sel << 3);
      A.i[v] = *(const int*)(A8 + (size_t)row * K + k0 + ab);
      const int bb = ((v >> 2) << 5) + (((v >> 1) & 1) << 4) + ((v & 1) << 2) + (sel << 3);
      B.i[v] = *(const int*)(bcol + k0 + bb);
    }
    acc = __builtin_amdgcn_wmma_i32_16x16x64_iu8(true, A.v, true, B.v, acc,
                                                 false, false);
  }

  const int col = n0 + rl;
  const int mbase = (mt << 4) + (sel << 3);
  if (mode == 0) {
    signed char* o = (signed char*)outp;
#pragma unroll
    for (int r = 0; r < 8; ++r)
      o[(size_t)(mbase + r) * N + col] =
          (acc[r] >= 0) ? (signed char)1 : (signed char)(-1);
  } else {
    _Float16* o = (_Float16*)outp;
#pragma unroll
    for (int r = 0; r < 8; ++r)
      o[(size_t)(mbase + r) * N + col] = (_Float16)(float)acc[r];
  }
}

__global__ void __launch_bounds__(128) head_wmma(
    const _Float16* __restrict__ h16, const _Float16* __restrict__ wout,
    const _Float16* __restrict__ bout, float* __restrict__ logits,
    int M, int K, int Ncols) {
  const int lane = threadIdx.x & 31;
  const int wave = threadIdx.x >> 5;
  const int tile = blockIdx.x * 4 + wave;
  const int mt = tile >> 1;
  const int nt = tile & 1;
  const int rl = lane & 15;
  const int sel = lane >> 4;
  const int row = (mt << 4) + rl;
  const int col = (nt << 4) + rl;
  const bool colvalid = col < Ncols;

  v8f acc = {};
  for (int k0 = 0; k0 < K; k0 += 32) {
    V16HB A, B;
#pragma unroll
    for (int v = 0; v < 8; ++v) {
      const int ab = ((v >> 2) << 4) + (sel << 3) + ((v & 3) << 1);
      A.u[v] = *(const unsigned*)(h16 + (size_t)row * K + k0 + ab);
      const int bb = ((v >> 2) << 4) + (sel << 3) + ((v & 3) << 1);
      B.u[v] = colvalid ? *(const unsigned*)(wout + (size_t)col * K + k0 + bb)
                        : 0u;
    }
    acc = __builtin_amdgcn_wmma_f32_16x16x32_f16(false, A.v, false, B.v,
                                                 (short)0, acc, false, false);
  }

  if (colvalid) {
    const float bias = (float)bout[col];
    const int mbase = (mt << 4) + (sel << 3);
#pragma unroll
    for (int r = 0; r < 8; ++r)
      logits[(size_t)(mbase + r) * Ncols + col] = acc[r] + bias;
  }
}

__global__ void logsoftmax_kernel(const float* __restrict__ logits,
                                  float* __restrict__ out, int cols) {
  const int row = blockIdx.x;
  const int t = threadIdx.x;
  __shared__ float s[32];
  __shared__ float mx, lse;
  const float v = (t < cols) ? logits[(size_t)row * cols + t] : -3.0e38f;
  s[t] = v;
  __syncthreads();
  if (t == 0) {
    float m = -3.0e38f;
    for (int i = 0; i < cols; ++i) m = fmaxf(m, s[i]);
    float sum = 0.0f;
    for (int i = 0; i < cols; ++i) sum += expf(s[i] - m);
    mx = m;
    lse = logf(sum);
  }
  __syncthreads();
  if (t < cols) out[(size_t)row * cols + t] = v - mx - lse;
}

extern "C" void kernel_launch(void* const* d_in, const int* in_sizes, int n_in,
                              void* d_out, int out_size, void* d_ws,
                              size_t ws_size, hipStream_t stream) {
  (void)in_sizes; (void)n_in; (void)out_size; (void)ws_size;

  const float* x = (const float*)d_in[0];
  const float* wconv[8];
  for (int i = 0; i < 8; ++i) wconv[i] = (const float*)d_in[1 + i];
  const float* wl1 = (const float*)d_in[9];
  const float* wl2 = (const float*)d_in[10];
  const _Float16* wout = (const _Float16*)d_in[11];
  const _Float16* bout = (const _Float16*)d_in[12];

  static const int CIN[8]  = {24, 64, 128, 256, 256, 512, 512, 512};
  static const int COUT[8] = {64, 128, 256, 256, 512, 512, 512, 512};
  static const bool POOL[8] = {true, true, false, true, false, true, false, true};

  char* ws = (char*)d_ws;
  size_t off = 0;
  auto carve = [&](size_t bytes) -> char* {
    char* p = ws + off;
    off += (bytes + 255) & ~(size_t)255;
    return p;
  };
  signed char* bufA = (signed char*)carve(512ull * 64 * 32 * 32);
  signed char* bufB = (signed char*)carve(512ull * 64 * 32 * 32);
  signed char* wsg[8];
  for (int i = 0; i < 8; ++i)
    wsg[i] = (signed char*)carve((size_t)COUT[i] * CIN[i] * 9);
  signed char* wl1s = (signed char*)carve(512 * 512);
  signed char* wl2s = (signed char*)carve(512 * 512);

  auto gsz = [](int n) {
    int b = (n + 255) / 256;
    return b > 4096 ? 4096 : b;
  };

  binarize_x_nhwc<<<dim3(gsz(512 * 24 * 32 * 32)), dim3(256), 0, stream>>>(
      x, bufA, 512, 24, 32, 32);
  for (int i = 0; i < 8; ++i) {
    const int n = COUT[i] * CIN[i] * 9;
    binarize_w_reorder<<<dim3(gsz(n)), dim3(256), 0, stream>>>(wconv[i], wsg[i],
                                                               COUT[i], CIN[i]);
  }
  binarize_f32_i8<<<dim3(gsz(512 * 512)), dim3(256), 0, stream>>>(wl1, wl1s,
                                                                  512 * 512);
  binarize_f32_i8<<<dim3(gsz(512 * 512)), dim3(256), 0, stream>>>(wl2, wl2s,
                                                                  512 * 512);

  signed char* cur = bufA;
  signed char* alt = bufB;
  int H = 32, W = 32;
  for (int i = 0; i < 8; ++i) {
    const int Mtiles = (512 * H * W) / 16;
    dim3 grid(Mtiles / 4, COUT[i] / 16);
    const size_t sh = (size_t)16 * CIN[i] * 9;
    switch (CIN[i]) {
      case 24:  conv_bin_wmma<24> <<<grid, dim3(128), sh, stream>>>(cur, wsg[i], alt, 512, H, W, COUT[i]); break;
      case 64:  conv_bin_wmma<64> <<<grid, dim3(128), sh, stream>>>(cur, wsg[i], alt, 512, H, W, COUT[i]); break;
      case 128: conv_bin_wmma<128><<<grid, dim3(128), sh, stream>>>(cur, wsg[i], alt, 512, H, W, COUT[i]); break;
      case 256: conv_bin_wmma<256><<<grid, dim3(128), sh, stream>>>(cur, wsg[i], alt, 512, H, W, COUT[i]); break;
      default:  conv_bin_wmma<512><<<grid, dim3(128), sh, stream>>>(cur, wsg[i], alt, 512, H, W, COUT[i]); break;
    }
    if (POOL[i]) {
      const int total = 512 * COUT[i] * (H / 2) * (W / 2);
      pool2_i8_nhwc<<<dim3(gsz(total)), dim3(256), 0, stream>>>(alt, cur, 512,
                                                                COUT[i], H, W);
      H >>= 1;
      W >>= 1;
    } else {
      signed char* t = cur; cur = alt; alt = t;
    }
  }

  gemm_bin_wmma<<<dim3(8, 32), dim3(128), 0, stream>>>(cur, wl1s, (void*)alt,
                                                       512, 512, 512,  0);
  _Float16* h16 = (_Float16*)cur;
  gemm_bin_wmma<<<dim3(8, 32), dim3(128), 0, stream>>>(alt, wl2s, (void*)h16,
                                                       512, 512, 512,  1);

  float* logits = (float*)alt;
  head_wmma<<<dim3(16), dim3(128), 0, stream>>>(h16, wout, bout, logits, 512,
                                                512, 18);

  logsoftmax_kernel<<<dim3(512), dim3(32), 0, stream>>>(logits, (float*)d_out, 18);
}
